// ChaoticAttentionMechanism_71683004170583
// MI455X (gfx1250) — matrix operand order repaired and checked by reading, not compiled, not run on hardware
//
#include <hip/hip_runtime.h>

typedef __bf16 bf16;
typedef __attribute__((ext_vector_type(16))) __bf16 v16bf;
typedef __attribute__((ext_vector_type(8)))  float  v8f;
typedef __attribute__((ext_vector_type(4)))  unsigned int u32x4;

union Frag16 { v16bf bf; u32x4 q[2]; };

#define DEVI static __device__ __forceinline__

namespace {
constexpr int  kB = 4, kS = 2048, kD = 1024, kHALF = 512;
constexpr long kM = (long)kB * kS;   // 8192 total rows (b*s flattened)
constexpr int  kKT = kS / 64;        // 32 column-tiles per row for sync stats
}

// ---------------------------------------------------------------- helpers

DEVI bf16 f2bf(float f) {
  unsigned u = __builtin_bit_cast(unsigned, f);
  unsigned r = u + 0x7FFFu + ((u >> 16) & 1u);   // round-to-nearest-even
  unsigned short h = (unsigned short)(r >> 16);
  return __builtin_bit_cast(bf16, h);
}

// A fragment (16x32, MxK) from row-major [M, ld] at (m0, k0).
// Lanes 0-15: row m0+r, K {0..15}; lanes 16-31: same rows, K {16..31}.
DEVI Frag16 load_a_frag(const bf16* base, int ld, int m0, int k0, int lane) {
  int half = lane >> 4, r = lane & 15;
  const bf16* p = base + (long)(m0 + r) * ld + k0 + half * 16;
  Frag16 f;
  f.q[0] = *(const u32x4*)(p);
  f.q[1] = *(const u32x4*)(p + 8);
  return f;
}

// B fragment (32x16, KxN) where B[k,n] = W[n,k], W row-major [N, ld].
// Lane (r,half): column n0+r, K = k0 + 16*half + 0..15 -> one contiguous 32B run.
DEVI Frag16 load_b_frag(const bf16* base, int ld, int n0, int k0, int lane) {
  int half = lane >> 4, r = lane & 15;
  const bf16* p = base + (long)(n0 + r) * ld + k0 + half * 16;
  Frag16 f;
  f.q[0] = *(const u32x4*)(p);
  f.q[1] = *(const u32x4*)(p + 8);
  return f;
}

DEVI v8f wmma_bf16(const Frag16& a, const Frag16& b, v8f c) {
  return __builtin_amdgcn_wmma_f32_16x16x32_bf16(
      /*neg_a=*/false, a.bf, /*neg_b=*/false, b.bf,
      /*c_mod=*/(short)0, c, /*reuse_a=*/false, /*reuse_b=*/false);
}

// ---------------------------------------------------------------- elementwise

__global__ __launch_bounds__(256)
void cvt_f32_bf16(const float* __restrict__ in, bf16* __restrict__ out, long n) {
  long i = (long)blockIdx.x * 256 + threadIdx.x;
  if (i < n) out[i] = f2bf(in[i]);
}

// pf[row, 0:512]=re, pf[row, 512:1024]=im  ->  U[row] = [re/h, im/h] (bf16)
__global__ __launch_bounds__(256)
void phase_kernel(const float* __restrict__ pf, bf16* __restrict__ U) {
  long i = (long)blockIdx.x * 256 + threadIdx.x;
  if (i >= kM * (long)kHALF) return;
  long row = i / kHALF;
  int  j   = (int)(i - row * kHALF);
  const float* p = pf + row * kD;
  float re = p[j], im = p[j + kHALF];
  float h2 = re * re + im * im;
  float c, s;
  if (h2 > 0.0f) { float rh = rsqrtf(h2); c = re * rh; s = im * rh; }
  else           { c = 1.0f; s = 0.0f; }            // atan2(0,0)=0
  bf16* u = U + row * kD;
  u[j] = f2bf(c);
  u[j + kHALF] = f2bf(s);
}

// [b, rows, cols] -> [b, cols, rows], LDS-tiled, bank-conflict padded
__global__ __launch_bounds__(256)
void transpose_bf16(const bf16* __restrict__ in, bf16* __restrict__ out,
                    int rows, int cols) {
  __shared__ bf16 tile[32][33];
  int b  = blockIdx.z;
  int c0 = blockIdx.x * 32, r0 = blockIdx.y * 32;
  const bf16* src = in + (long)b * rows * cols;
  bf16* dst = out + (long)b * rows * cols;
  int tx = threadIdx.x & 31;
  int ty = threadIdx.x >> 5;           // 0..7
  #pragma unroll
  for (int rr = ty; rr < 32; rr += 8)
    tile[rr][tx] = src[(long)(r0 + rr) * cols + c0 + tx];
  __syncthreads();
  #pragma unroll
  for (int cc = ty; cc < 32; cc += 8)
    dst[(long)(c0 + cc) * rows + r0 + tx] = tile[tx][cc];
}

// ---------------------------------------------------------------- generic GEMM
// C[m,n] = sum_k A[m,k] * W[n,k] (+ bias[n]).  A:[M,K] bf16, W:[N,K] bf16.
// Wave computes 64x64 via 4x4 WMMA fragments; block = 8 waves = 128x256 tile.
__global__ __launch_bounds__(256)
void gemm_bf16_wt(const bf16* __restrict__ A, const bf16* __restrict__ W,
                  const float* __restrict__ bias,
                  float* __restrict__ Cf, bf16* __restrict__ Ch,
                  int K, int N,
                  long strideA, long strideW, long strideC) {
  int lane = threadIdx.x & 31;
  int wave = threadIdx.x >> 5;                  // 0..7
  int z = blockIdx.z;
  const bf16* Ab = A + (long)z * strideA;
  const bf16* Wb = W + (long)z * strideW;
  int m0 = blockIdx.y * 128 + (wave >> 2) * 64;
  int n0 = blockIdx.x * 256 + (wave & 3) * 64;

  v8f acc[4][4] = {};
  int r16 = lane & 15;

  #pragma unroll 1
  for (int k0 = 0; k0 < K; k0 += 32) {
    if (k0 + 32 < K) {                          // global_prefetch_b8 flavor
      __builtin_prefetch(Ab + (long)(m0 + r16) * K + k0 + 32, 0, 3);
      __builtin_prefetch(Wb + (long)(n0 + r16) * K + k0 + 32, 0, 3);
    }
    Frag16 af[4], bf_[4];
    #pragma unroll
    for (int i = 0; i < 4; ++i) af[i]  = load_a_frag(Ab, K, m0 + 16 * i, k0, lane);
    #pragma unroll
    for (int j = 0; j < 4; ++j) bf_[j] = load_b_frag(Wb, K, n0 + 16 * j, k0, lane);
    #pragma unroll
    for (int i = 0; i < 4; ++i)
      #pragma unroll
      for (int j = 0; j < 4; ++j)
        acc[i][j] = wmma_bf16(af[i], bf_[j], acc[i][j]);
  }

  int half = lane >> 4;
  #pragma unroll
  for (int j = 0; j < 4; ++j) {
    int n = n0 + 16 * j + r16;
    float bv = bias ? bias[n] : 0.0f;
    #pragma unroll
    for (int i = 0; i < 4; ++i) {
      int mbase = m0 + 16 * i + half * 8;
      #pragma unroll
      for (int p = 0; p < 8; ++p) {
        float v = acc[i][j][p] + bv;
        long off = (long)z * strideC + (long)(mbase + p) * N + n;
        if (Cf) Cf[off] = v;
        if (Ch) Ch[off] = f2bf(v);
      }
    }
  }
}

// ---------------------------------------------------------------- fused scores
// Per wave: 32(q) x 64(k) tile of BOTH  s = Q.K^T/32  and  y = Uq.Uk^T/512,
// one shared K-loop (K = 1024 for both). Epilogue: chaotic = s + pc*y + bp*t(1-t),
// plus deterministic per-(row, 64-col-tile) sync sum / sumsq for the variance loss.
__global__ __launch_bounds__(128)
void score_kernel(const bf16* __restrict__ Qh, const bf16* __restrict__ Kh,
                  const bf16* __restrict__ Uq, const bf16* __restrict__ Uk,
                  float* __restrict__ chaotic,
                  float* __restrict__ psum, float* __restrict__ psumsq,
                  const float* __restrict__ bifp, const float* __restrict__ pcp) {
  int lane = threadIdx.x & 31;
  int wave = threadIdx.x >> 5;                    // 0..3
  int z  = blockIdx.z;
  int q0 = blockIdx.y * 64  + (wave >> 1) * 32;
  int n0 = blockIdx.x * 128 + (wave & 1) * 64;

  const bf16* Q  = Qh + (long)z * kS * kD;
  const bf16* Kp = Kh + (long)z * kS * kD;
  const bf16* UA = Uq + (long)z * kS * kD;
  const bf16* UB = Uk + (long)z * kS * kD;

  v8f as_[2][4] = {};
  v8f ay_[2][4] = {};

  #pragma unroll 1
  for (int k0 = 0; k0 < kD; k0 += 32) {
    Frag16 aq[2], au[2], bk[4], bu[4];
    #pragma unroll
    for (int i = 0; i < 2; ++i) {
      aq[i] = load_a_frag(Q,  kD, q0 + 16 * i, k0, lane);
      au[i] = load_a_frag(UA, kD, q0 + 16 * i, k0, lane);
    }
    #pragma unroll
    for (int j = 0; j < 4; ++j) {
      bk[j] = load_b_frag(Kp, kD, n0 + 16 * j, k0, lane);
      bu[j] = load_b_frag(UB, kD, n0 + 16 * j, k0, lane);
    }
    #pragma unroll
    for (int i = 0; i < 2; ++i)
      #pragma unroll
      for (int j = 0; j < 4; ++j) {
        as_[i][j] = wmma_bf16(aq[i], bk[j], as_[i][j]);
        ay_[i][j] = wmma_bf16(au[i], bu[j], ay_[i][j]);
      }
  }

  float bparam = bifp[0];
  float pcoup  = pcp[0];
  const float inv_sqrt_hd = 0.03125f;     // 1/sqrt(1024)
  const float inv_half    = 1.0f / 512.0f;
  int half = lane >> 4, r = lane & 15;
  int kt = blockIdx.x * 2 + (wave & 1);   // which 64-wide column tile (0..31)

  #pragma unroll
  for (int i = 0; i < 2; ++i) {
    #pragma unroll
    for (int p = 0; p < 8; ++p) {
      int m = q0 + 16 * i + half * 8 + p;
      float ys = 0.0f, yss = 0.0f;
      #pragma unroll
      for (int j = 0; j < 4; ++j) {
        float s = as_[i][j][p] * inv_sqrt_hd;
        float y = ay_[i][j][p] * inv_half;
        float t = tanhf(s);
        float ch = s + pcoup * y + bparam * t * (1.0f - t);
        chaotic[((long)z * kS + m) * kS + (n0 + 16 * j + r)] = ch;
        ys  += y;
        yss += y * y;
      }
      // reduce 16 lanes within each half-wave (rows stay separate)
      #pragma unroll
      for (int d2 = 1; d2 < 16; d2 <<= 1) {
        ys  += __shfl_xor(ys,  d2, 32);
        yss += __shfl_xor(yss, d2, 32);
      }
      if (r == 0) {
        long sidx = ((long)z * kS + m) * kKT + kt;   // unique writer
        psum[sidx]   = ys;
        psumsq[sidx] = yss;
      }
    }
  }
}

// ---------------------------------------------------------------- softmax rows
__global__ __launch_bounds__(256)
void softmax_kernel(const float* __restrict__ ch, bf16* __restrict__ attn) {
  long row = blockIdx.x;
  const float* src = ch + row * kS;
  bf16* dst = attn + row * kS;
  __shared__ float red[8];
  int tid = threadIdx.x, lane = tid & 31, wv = tid >> 5;

  float m = -3.4e38f;
  for (int j = tid; j < kS; j += 256) m = fmaxf(m, src[j]);
  #pragma unroll
  for (int d2 = 1; d2 < 32; d2 <<= 1) m = fmaxf(m, __shfl_xor(m, d2, 32));
  if (lane == 0) red[wv] = m;
  __syncthreads();
  if (wv == 0) {
    float v = (lane < 8) ? red[lane] : -3.4e38f;
    #pragma unroll
    for (int d2 = 1; d2 < 8; d2 <<= 1) v = fmaxf(v, __shfl_xor(v, d2, 32));
    if (lane == 0) red[0] = v;
  }
  __syncthreads();
  m = red[0];
  __syncthreads();

  float sum = 0.0f;
  for (int j = tid; j < kS; j += 256) sum += expf(src[j] - m);
  #pragma unroll
  for (int d2 = 1; d2 < 32; d2 <<= 1) sum += __shfl_xor(sum, d2, 32);
  if (lane == 0) red[wv] = sum;
  __syncthreads();
  if (wv == 0) {
    float v = (lane < 8) ? red[lane] : 0.0f;
    #pragma unroll
    for (int d2 = 1; d2 < 8; d2 <<= 1) v += __shfl_xor(v, d2, 32);
    if (lane == 0) red[0] = v;
  }
  __syncthreads();
  float inv = 1.0f / red[0];
  for (int j = tid; j < kS; j += 256) dst[j] = f2bf(expf(src[j] - m) * inv);
}

// ---------------------------------------------------------------- sync loss
// Single block -> fully deterministic reduction.
// loss = 0.01 * mean_rows( (sumsq - sum^2/S) / (S-1) )
__global__ __launch_bounds__(256)
void loss_kernel(const float* __restrict__ psum, const float* __restrict__ psumsq,
                 float* __restrict__ loss) {
  __shared__ float red[8];
  int tid = threadIdx.x, lane = tid & 31, wv = tid >> 5;
  float accv = 0.0f;
  const float Nf = (float)kS;
  for (long rowi = tid; rowi < kM; rowi += 256) {
    float s = 0.0f, ss = 0.0f;
    #pragma unroll
    for (int t = 0; t < kKT; ++t) {
      s  += psum[rowi * kKT + t];
      ss += psumsq[rowi * kKT + t];
    }
    accv += (ss - s * s / Nf) / (Nf - 1.0f);
  }
  #pragma unroll
  for (int d2 = 1; d2 < 32; d2 <<= 1) accv += __shfl_xor(accv, d2, 32);
  if (lane == 0) red[wv] = accv;
  __syncthreads();
  if (tid == 0) {
    float t = 0.0f;
    #pragma unroll
    for (int i = 0; i < 8; ++i) t += red[i];
    *loss = 0.01f * t / (float)kM;
  }
}

// ---------------------------------------------------------------- launch

extern "C" void kernel_launch(void* const* d_in, const int* in_sizes, int n_in,
                              void* d_out, int out_size, void* d_ws, size_t ws_size,
                              hipStream_t stream) {
  (void)in_sizes; (void)n_in; (void)out_size; (void)ws_size;
  const float* x   = (const float*)d_in[0];
  const float* Wq  = (const float*)d_in[1];
  const float* bq  = (const float*)d_in[2];
  const float* Wk  = (const float*)d_in[3];
  const float* bk  = (const float*)d_in[4];
  const float* Wv  = (const float*)d_in[5];
  const float* bv  = (const float*)d_in[6];
  const float* Wp  = (const float*)d_in[7];
  const float* bp  = (const float*)d_in[8];
  const float* Wo  = (const float*)d_in[9];
  const float* bo  = (const float*)d_in[10];
  const float* bif = (const float*)d_in[11];
  const float* pc  = (const float*)d_in[12];

  char* ws = (char*)d_ws;
  size_t off = 0;
  auto alloc = [&](size_t bytes) -> void* {
    void* p = ws + off;
    off += (bytes + 255) & ~(size_t)255;
    return p;
  };
  const size_t actBytes = (size_t)kM * kD * 2;       // 16 MiB (bf16 [8192,1024])
  const size_t wBytes   = (size_t)kD * kD * 2;       // 2 MiB
  bf16* xh   = (bf16*)alloc(actBytes);
  bf16* Wqh  = (bf16*)alloc(wBytes);
  bf16* Wkh  = (bf16*)alloc(wBytes);
  bf16* Wvh  = (bf16*)alloc(wBytes);
  bf16* Wph  = (bf16*)alloc(wBytes);
  bf16* Woh  = (bf16*)alloc(wBytes);
  bf16* Qh   = (bf16*)alloc(actBytes);
  bf16* Kh   = (bf16*)alloc(actBytes);
  bf16* Vh   = (bf16*)alloc(actBytes);
  bf16* Vt   = (bf16*)alloc(actBytes);               // [b, d, s]
  float* pf  = (float*)alloc((size_t)kM * kD * 4);   // reused for pfQ then pfK
  bf16* Uq   = (bf16*)alloc(actBytes);
  bf16* Uk   = (bf16*)alloc(actBytes);
  float* cha = (float*)alloc((size_t)kB * kS * kS * 4);   // 64 MiB
  bf16* attn = (bf16*)alloc((size_t)kB * kS * kS * 2);    // 32 MiB
  bf16* avh  = (bf16*)alloc(actBytes);
  float* psum   = (float*)alloc((size_t)kM * kKT * 4);
  float* psumsq = (float*)alloc((size_t)kM * kKT * 4);

  float* outp = (float*)d_out;
  float* lossp = outp + kM * (long)kD;

  // 1) fp32 -> bf16 conversions
  long nAct = kM * (long)kD, nW = (long)kD * kD;
  cvt_f32_bf16<<<dim3((unsigned)((nAct + 255) / 256)), 256, 0, stream>>>(x, xh, nAct);
  cvt_f32_bf16<<<dim3((unsigned)((nW + 255) / 256)), 256, 0, stream>>>(Wq, Wqh, nW);
  cvt_f32_bf16<<<dim3((unsigned)((nW + 255) / 256)), 256, 0, stream>>>(Wk, Wkh, nW);
  cvt_f32_bf16<<<dim3((unsigned)((nW + 255) / 256)), 256, 0, stream>>>(Wv, Wvh, nW);
  cvt_f32_bf16<<<dim3((unsigned)((nW + 255) / 256)), 256, 0, stream>>>(Wp, Wph, nW);
  cvt_f32_bf16<<<dim3((unsigned)((nW + 255) / 256)), 256, 0, stream>>>(Wo, Woh, nW);

  // 2) Q/K/V projections: [8192,1024] x [1024,1024]^T
  dim3 gProj(kD / 256, (unsigned)(kM / 128), 1);
  gemm_bf16_wt<<<gProj, 256, 0, stream>>>(xh, Wqh, bq, nullptr, Qh, kD, kD, 0, 0, 0);
  gemm_bf16_wt<<<gProj, 256, 0, stream>>>(xh, Wkh, bk, nullptr, Kh, kD, kD, 0, 0, 0);
  gemm_bf16_wt<<<gProj, 256, 0, stream>>>(xh, Wvh, bv, nullptr, Vh, kD, kD, 0, 0, 0);

  // 3) phase unit vectors Uq, Uk (pf buffer reused sequentially)
  unsigned gPh = (unsigned)((kM * (long)kHALF + 255) / 256);
  gemm_bf16_wt<<<gProj, 256, 0, stream>>>(Qh, Wph, bp, pf, nullptr, kD, kD, 0, 0, 0);
  phase_kernel<<<gPh, 256, 0, stream>>>(pf, Uq);
  gemm_bf16_wt<<<gProj, 256, 0, stream>>>(Kh, Wph, bp, pf, nullptr, kD, kD, 0, 0, 0);
  phase_kernel<<<gPh, 256, 0, stream>>>(pf, Uk);

  // 4) V^T per batch (for attn @ V as A @ W^T)
  transpose_bf16<<<dim3(kD / 32, kS / 32, kB), 256, 0, stream>>>(Vh, Vt, kS, kD);

  // 5) fused chaotic scores (QK^T and UqUk^T in one K-loop) + sync stats
  score_kernel<<<dim3(kS / 128, kS / 64, kB), 128, 0, stream>>>(
      Qh, Kh, Uq, Uk, cha, psum, psumsq, bif, pc);

  // 6) softmax rows -> bf16 attn
  softmax_kernel<<<dim3((unsigned)kM), 256, 0, stream>>>(cha, attn);

  // 7) av = attn @ Vt^T  (batched: [2048,2048] x [1024,2048]^T)
  dim3 gAV(kD / 256, kS / 128, kB);
  gemm_bf16_wt<<<gAV, 256, 0, stream>>>(attn, Vt, nullptr, nullptr, avh, kS, kD,
                                        (long)kS * kS, (long)kD * kS, (long)kS * kD);

  // 8) out = av @ Wo^T + bo  (fp32 straight into d_out)
  gemm_bf16_wt<<<gProj, 256, 0, stream>>>(avh, Woh, bo, outp, nullptr, kD, kD, 0, 0, 0);

  // 9) sync variance loss -> d_out[last]
  loss_kernel<<<1, 256, 0, stream>>>(psum, psumsq, lossp);
}
